// BridgeEncoder_12584254177962
// MI455X (gfx1250) — matrix operand order repaired and checked by reading, not compiled, not run on hardware
//
#include <hip/hip_runtime.h>
#include <stdint.h>

// ---------------------------------------------------------------------------
// Problem constants (reference: x (4,4096,1024) f32, W (4096,1024), b (4096))
// ---------------------------------------------------------------------------
#define D_DENSE  1024
#define D_SPARSE 4096
#define TOPK     1024

#define BM 128
#define BN 128
#define BK 32
#define KT (D_DENSE / BK)   // 32 K-steps

typedef __attribute__((ext_vector_type(16))) __bf16       v16bf;
typedef __attribute__((ext_vector_type(8)))  float        v8f;
typedef __attribute__((ext_vector_type(4)))  unsigned int u32x4;
typedef __attribute__((ext_vector_type(8)))  int          i32x8;
typedef __attribute__((ext_vector_type(4)))  int          i32x4;

// ---------------------------------------------------------------------------
// TDM: build a 2D-tile D# (groups 0/1 per CDNA5 ISA 8.3/8.4) and issue
// tensor_load_to_lds.  data_size=4B, row-major tensor, tile = tile_d0 x tile_d1
// copied contiguously into LDS (=> LDS layout [tile_d1][tile_d0] f32).
// ---------------------------------------------------------------------------
__device__ __forceinline__ unsigned lds_off_of(const void* p) {
  // low 32 bits of a generic pointer into the shared aperture == LDS byte addr
  return (unsigned)(uintptr_t)p;
}

__device__ __forceinline__ void tdm_load_tile_2d(unsigned lds_off,
                                                 const float* gptr,
                                                 unsigned tensor_d0,   // elems
                                                 unsigned tensor_d1,   // rows
                                                 unsigned stride0,     // elems
                                                 unsigned tile_d0,
                                                 unsigned tile_d1) {
  unsigned long long ga = (unsigned long long)(uintptr_t)gptr;
  u32x4 g0;
  g0[0] = 1u;                                              // count=1, user mode
  g0[1] = lds_off;                                         // lds_addr [63:32]
  g0[2] = (unsigned)(ga & 0xFFFFFFFFull);                  // global_addr lo
  g0[3] = (unsigned)((ga >> 32) & 0x01FFFFFFull)           // global_addr[56:32]
        | (2u << 30);                                      // type=2 ("image")
  i32x8 g1;
  g1[0] = (int)(2u << 16);                                 // data_size=2 (4B)
  g1[1] = (int)((tensor_d0 & 0xFFFFu) << 16);              // tensor_dim0 lo16
  g1[2] = (int)(((tensor_d0 >> 16) & 0xFFFFu)              // tensor_dim0 hi16
        |        ((tensor_d1 & 0xFFFFu) << 16));           // tensor_dim1 lo16
  g1[3] = (int)(((tensor_d1 >> 16) & 0xFFFFu)              // tensor_dim1 hi16
        |        ((tile_d0 & 0xFFFFu) << 16));             // tile_dim0
  g1[4] = (int)(tile_d1 & 0xFFFFu);                        // tile_dim1 (dim2=0)
  g1[5] = (int)stride0;                                    // dim0_stride lo32
  g1[6] = 0;                                               // stride hi / dim1_stride
  g1[7] = 0;
  i32x4 z4 = (i32x4)0;
#if defined(__clang_major__) && (__clang_major__ >= 23)
  i32x8 z8 = (i32x8)0;
  __builtin_amdgcn_tensor_load_to_lds(g0, g1, z4, z4, z8, 0);
#else
  __builtin_amdgcn_tensor_load_to_lds(g0, g1, z4, z4, 0);
#endif
}

// pack 8 f32 -> bf16 elements [base..base+7] of a v16bf fragment
__device__ __forceinline__ void pack8(v16bf& d, int base, float4 a, float4 b) {
  d[base + 0] = (__bf16)a.x; d[base + 1] = (__bf16)a.y;
  d[base + 2] = (__bf16)a.z; d[base + 3] = (__bf16)a.w;
  d[base + 4] = (__bf16)b.x; d[base + 5] = (__bf16)b.y;
  d[base + 6] = (__bf16)b.z; d[base + 7] = (__bf16)b.w;
}

// ---------------------------------------------------------------------------
// Kernel 1: h = x * W^T + b   (bf16 WMMA, f32 accumulate) -> d_out
// grid  = (N/128, M/128), block = 256 threads = 8 waves (4x2 wave grid)
// ---------------------------------------------------------------------------
__global__ __launch_bounds__(256) void gemm_bias_kernel(
    const float* __restrict__ x, const float* __restrict__ W,
    const float* __restrict__ bias, float* __restrict__ out, int M) {
  __shared__ float lA[2][BM * BK];   // [row][k] f32, 16KB each buffer
  __shared__ float lB[2][BN * BK];   // [col(n)][k] f32 (== W sub-tile layout)

  const int tid  = threadIdx.x;
  const int lane = tid & 31;
  const int wid  = tid >> 5;     // 0..7
  const int wm   = wid >> 1;     // 0..3 : 32-row strip
  const int wn   = wid & 1;      // 0..1 : 64-col strip
  const int lr   = lane & 15;
  const int hi   = lane >> 4;    // half-wave select

  const int row0 = blockIdx.y * BM;
  const int col0 = blockIdx.x * BN;

  v8f acc[2][4];
#pragma unroll
  for (int mt = 0; mt < 2; ++mt)
#pragma unroll
    for (int nt = 0; nt < 4; ++nt) acc[mt][nt] = (v8f)0.0f;

  // prologue: stage K-step 0
  if (wid == 0) {
    tdm_load_tile_2d(lds_off_of(&lA[0][0]), x + (size_t)row0 * D_DENSE,
                     D_DENSE, (unsigned)M, D_DENSE, BK, BM);
    tdm_load_tile_2d(lds_off_of(&lB[0][0]), W + (size_t)col0 * D_DENSE,
                     D_DENSE, D_SPARSE, D_DENSE, BK, BN);
  }

  for (int kt = 0; kt < KT; ++kt) {
    const int cur = kt & 1;
    if (wid == 0) __builtin_amdgcn_s_wait_tensorcnt(0);
    __syncthreads();                      // buffer `cur` ready for everyone

    if (wid == 0 && (kt + 1) < KT) {      // prefetch next K-step into other buf
      const int k1 = (kt + 1) * BK;
      tdm_load_tile_2d(lds_off_of(&lA[1 - cur][0]),
                       x + (size_t)row0 * D_DENSE + k1, D_DENSE, (unsigned)M,
                       D_DENSE, BK, BM);
      tdm_load_tile_2d(lds_off_of(&lB[1 - cur][0]),
                       W + (size_t)col0 * D_DENSE + k1, D_DENSE, D_SPARSE,
                       D_DENSE, BK, BN);
    }

    // ---- assemble bf16 fragments from LDS (ISA 7.12.2 16-bit layouts) ----
    v16bf afrag[2];
#pragma unroll
    for (int mt = 0; mt < 2; ++mt) {
      const float* src = &lA[cur][(size_t)(wm * 32 + mt * 16 + lr) * BK];
      // lanes 0-15: K 0..7 then 16..23 ; lanes 16-31: K 8..15 then 24..31
      const float4* p0 = (const float4*)(src + hi * 8);
      const float4* p1 = (const float4*)(src + 16 + hi * 8);
      float4 q0 = p0[0], q1 = p0[1], q2 = p1[0], q3 = p1[1];
      pack8(afrag[mt], 0, q0, q1);
      pack8(afrag[mt], 8, q2, q3);
    }
    v16bf bfrag[4];
#pragma unroll
    for (int nt = 0; nt < 4; ++nt) {
      // B[k][n] = W[col][k]: lane n = lr, K hi*8..+8 then 16+hi*8..+8
      const float* src =
          &lB[cur][(size_t)(wn * 64 + nt * 16 + lr) * BK + hi * 8];
      const float4* p = (const float4*)src;
      float4 r0 = p[0], r1 = p[1], r2 = p[4], r3 = p[5];
      pack8(bfrag[nt], 0, r0, r1);
      pack8(bfrag[nt], 8, r2, r3);
    }

    // ---- 8 WMMAs: 32x64 tile per wave, K+=32 ----
#pragma unroll
    for (int mt = 0; mt < 2; ++mt)
#pragma unroll
      for (int nt = 0; nt < 4; ++nt)
        acc[mt][nt] = __builtin_amdgcn_wmma_f32_16x16x32_bf16(
            false, afrag[mt], false, bfrag[nt], (short)0, acc[mt][nt], false,
            false);

    __syncthreads();   // done reading buffer `cur` before it is re-staged
  }

  // ---- epilogue: add bias, store (C layout: lane n=lr, VGPR g -> M=hi*8+g)
#pragma unroll
  for (int mt = 0; mt < 2; ++mt) {
    const int m_base = row0 + wm * 32 + mt * 16 + hi * 8;
#pragma unroll
    for (int nt = 0; nt < 4; ++nt) {
      const int n = col0 + wn * 64 + nt * 16 + lr;
      const float bv = bias[n];
      v8f c = acc[mt][nt];
#pragma unroll
      for (int g = 0; g < 8; ++g)
        out[(size_t)(m_base + g) * D_SPARSE + n] = c[g] + bv;
    }
  }
}

// ---------------------------------------------------------------------------
// Kernel 2: exact per-row abs-top-k mask (radix select on |v| bit pattern).
// One 256-thread block per row of 4096; operates on d_out in place.
// ---------------------------------------------------------------------------
__global__ __launch_bounds__(256) void topk_mask_kernel(float* __restrict__ out) {
  __shared__ float    v[D_SPARSE];
  __shared__ unsigned hist[256];
  __shared__ unsigned sel_prefix, sel_remaining, tie_cnt;

  float* rp = out + (size_t)blockIdx.x * D_SPARSE;
  for (int i = threadIdx.x; i < D_SPARSE; i += 256) v[i] = rp[i];
  if (threadIdx.x == 0) { sel_prefix = 0u; sel_remaining = TOPK; tie_cnt = 0u; }
  __syncthreads();

  // 4 passes of 8-bit radix select (descending) on key = bits(|v|), monotone.
#pragma unroll
  for (int pass = 0; pass < 4; ++pass) {
    const int shift = 24 - 8 * pass;
    hist[threadIdx.x] = 0u;
    __syncthreads();
    const unsigned mask = pass ? (0xFFFFFFFFu << (shift + 8)) : 0u;
    const unsigned pfx  = sel_prefix;
    for (int i = threadIdx.x; i < D_SPARSE; i += 256) {
      unsigned u = __float_as_uint(fabsf(v[i]));
      if ((u & mask) == pfx) atomicAdd(&hist[(u >> shift) & 255u], 1u);
    }
    __syncthreads();
    if (threadIdx.x == 0) {
      unsigned need = sel_remaining, cum = 0u;
      int b = 255;
      for (; b > 0; --b) {
        cum += hist[b];
        if (cum >= need) break;
      }
      if (cum < need) cum += hist[0];                 // b==0 fallthrough
      sel_remaining = need - (cum - hist[b]);         // still needed inside bin
      sel_prefix    = pfx | ((unsigned)b << shift);
    }
    __syncthreads();
  }

  const unsigned T      = sel_prefix;     // key of k-th largest |v|
  const unsigned keepEq = sel_remaining;  // #elements equal to T to keep

  for (int i = threadIdx.x; i < D_SPARSE; i += 256) {
    unsigned u = __float_as_uint(fabsf(v[i]));
    float r = 0.0f;
    if (u > T) {
      r = v[i];
    } else if (u == T) {
      if (atomicAdd(&tie_cnt, 1u) < keepEq) r = v[i];
    }
    rp[i] = r;
  }
}

// ---------------------------------------------------------------------------
// Entry point
// ---------------------------------------------------------------------------
extern "C" void kernel_launch(void* const* d_in, const int* in_sizes, int n_in,
                              void* d_out, int out_size, void* d_ws,
                              size_t ws_size, hipStream_t stream) {
  (void)n_in; (void)out_size; (void)d_ws; (void)ws_size;
  const float* x  = (const float*)d_in[0];
  const float* W  = (const float*)d_in[1];
  const float* bv = (const float*)d_in[2];
  float* out = (float*)d_out;

  const int M = in_sizes[0] / D_DENSE;   // 4*4096 = 16384 rows

  dim3 grid1(D_SPARSE / BN, M / BM);     // (32, 128)
  gemm_bias_kernel<<<grid1, dim3(256), 0, stream>>>(x, W, bv, out, M);

  topk_mask_kernel<<<dim3(M), dim3(256), 0, stream>>>(out);
}
